// Gated_33337536151795
// MI455X (gfx1250) — compile-verified, matrix operand order checked by reading, not run on hardware
//
#include <hip/hip_runtime.h>
#include <hip/hip_bf16.h>

// ---------------------------------------------------------------------------
// GatedGraphConv pipeline for MI455X (gfx1250, wave32, WMMA).
// All dense GEMMs use v_wmma_f32_16x16x32_bf16 (f32 accum, bf16 operands,
// native v_cvt bf16 conversion). Scatter-add runs as fp32 global atomics
// (agg = 51MB fits in 192MB L2). Classifier stages w2 via async LDS copy.
// ---------------------------------------------------------------------------

typedef __attribute__((ext_vector_type(16))) __bf16 v16bf;
typedef __attribute__((ext_vector_type(8)))  float  v8f;

#define HID    64
#define NFEAT  256
#define NCLASS 40

// Build a 16x32 (A) or 32x16 (B) bf16 fragment for one lane using native
// hardware converts. p points at element K = k0 + 8*(lane>=16) along the K
// axis; `stride` is the element stride along K.
// elems[0..7] = K..K+7, elems[8..15] = K+16..K+23 (ISA 16-bit A/B layout).
__device__ __forceinline__ v16bf make_frag(const float* __restrict__ p, int stride) {
    v16bf f;
#pragma unroll
    for (int j = 0; j < 8; ++j) {
        f[j]     = (__bf16)p[j * stride];
        f[j + 8] = (__bf16)p[(j + 16) * stride];
    }
    return f;
}

// ---------------------------------------------------------------------------
// Kernel 1: h = relu(x @ w1.T + b1)   x:[N,256] w1:[64,256] -> h:[N,64]
// block = 256 threads = 8 waves; each wave: 16 rows x 64 cols (4 col tiles).
// ---------------------------------------------------------------------------
__global__ __launch_bounds__(256)
void k_input_gemm(const float* __restrict__ x, const float* __restrict__ w1,
                  const float* __restrict__ b1, float* __restrict__ h, int N) {
    const int lane = threadIdx.x & 31;
    const int wave = threadIdx.x >> 5;
    const int rowbase = blockIdx.x * 128 + wave * 16;
    const int arow = rowbase + (lane & 15);
    const int arow_c = (arow < N) ? arow : (N - 1);
    const int koff = (lane >> 4) * 8;
    const int col  = lane & 15;

    v8f acc[4] = {};
    for (int k0 = 0; k0 < NFEAT; k0 += 32) {
        v16bf a = make_frag(x + (size_t)arow_c * NFEAT + k0 + koff, 1);
#pragma unroll
        for (int t = 0; t < 4; ++t) {
            // B[k][n] = w1[n][k]  (row of w1 is contiguous in K)
            v16bf b = make_frag(w1 + (size_t)(t * 16 + col) * NFEAT + k0 + koff, 1);
            acc[t] = __builtin_amdgcn_wmma_f32_16x16x32_bf16(
                false, a, false, b, (short)0, acc[t], false, false);
        }
    }
    const int r0 = (lane >> 4) * 8;
#pragma unroll
    for (int t = 0; t < 4; ++t) {
        float bias = b1[t * 16 + col];
#pragma unroll
        for (int r = 0; r < 8; ++r) {
            int row = rowbase + r0 + r;
            if (row < N) {
                float v = acc[t][r] + bias;
                h[(size_t)row * HID + t * 16 + col] = v > 0.f ? v : 0.f;
            }
        }
    }
}

// ---------------------------------------------------------------------------
// Kernel 2: m = h @ W   h:[N,64] W:[64,64] (not transposed) -> m:[N,64]
// ---------------------------------------------------------------------------
__global__ __launch_bounds__(256)
void k_msg_gemm(const float* __restrict__ h, const float* __restrict__ W,
                float* __restrict__ m, int N) {
    const int lane = threadIdx.x & 31;
    const int wave = threadIdx.x >> 5;
    const int rowbase = blockIdx.x * 128 + wave * 16;
    const int arow = rowbase + (lane & 15);
    const int arow_c = (arow < N) ? arow : (N - 1);
    const int koff = (lane >> 4) * 8;
    const int col  = lane & 15;

    v8f acc[4] = {};
#pragma unroll
    for (int k0 = 0; k0 < HID; k0 += 32) {
        v16bf a = make_frag(h + (size_t)arow_c * HID + k0 + koff, 1);
#pragma unroll
        for (int t = 0; t < 4; ++t) {
            // B[k][n] = W[k*64 + n] : stride 64 along K
            v16bf b = make_frag(W + (size_t)(k0 + koff) * HID + (t * 16 + col), HID);
            acc[t] = __builtin_amdgcn_wmma_f32_16x16x32_bf16(
                false, a, false, b, (short)0, acc[t], false, false);
        }
    }
    const int r0 = (lane >> 4) * 8;
#pragma unroll
    for (int t = 0; t < 4; ++t) {
#pragma unroll
        for (int r = 0; r < 8; ++r) {
            int row = rowbase + r0 + r;
            if (row < N)
                m[(size_t)row * HID + t * 16 + col] = acc[t][r];
        }
    }
}

// ---------------------------------------------------------------------------
// Kernel 3: scatter-add  agg[dst[e], :] += m[src[e], :]
// 64 lanes per edge; fp32 atomics resolve in L2 (agg = 51MB << 192MB L2).
// Prefetch the index stream ahead (global_prefetch_b8).
// ---------------------------------------------------------------------------
__global__ __launch_bounds__(256)
void k_scatter(const float* __restrict__ m, const int* __restrict__ src,
               const int* __restrict__ dst, float* __restrict__ agg, int E) {
    long long gid = (long long)blockIdx.x * blockDim.x + threadIdx.x;
    int e = (int)(gid >> 6);
    int c = (int)(gid & 63);
    if (e < E) {
        if (e + 2048 < E) {
            __builtin_prefetch(src + e + 2048, 0, 0);
            __builtin_prefetch(dst + e + 2048, 0, 0);
        }
        int s = src[e];
        int d = dst[e];
        atomicAdd(agg + (size_t)d * HID + c, m[(size_t)s * HID + c]);
    }
}

// ---------------------------------------------------------------------------
// Kernel 4: GRU cell. Per block: 16 nodes.
// gi = agg @ w_ih.T, gh = h @ w_hh.T, both [16,192] via WMMA into LDS,
// then fused gate math (sigmoid/tanh) updates h in place.
// 24 output tiles (2 mats x 12 col tiles) over 8 waves -> 3 tiles per wave.
// ---------------------------------------------------------------------------
__global__ __launch_bounds__(256)
void k_gru(const float* __restrict__ agg, float* __restrict__ h,
           const float* __restrict__ w_ih, const float* __restrict__ w_hh,
           const float* __restrict__ b_ih, const float* __restrict__ b_hh, int N) {
    __shared__ float gi[16 * 192];
    __shared__ float gh[16 * 192];

    const int lane = threadIdx.x & 31;
    const int wave = threadIdx.x >> 5;
    const int nodebase = blockIdx.x * 16;
    const int arow = nodebase + (lane & 15);
    const int arow_c = (arow < N) ? arow : (N - 1);
    const int koff = (lane >> 4) * 8;
    const int col  = lane & 15;
    const int r0   = (lane >> 4) * 8;

#pragma unroll
    for (int i = 0; i < 3; ++i) {
        int t = wave * 3 + i;            // 0..23
        int matsel = t / 12;             // 0 -> gi (agg,w_ih), 1 -> gh (h,w_hh)
        int ct = t % 12;                 // column tile of 192
        const float* A  = matsel ? h    : agg;
        const float* Wt = matsel ? w_hh : w_ih;
        v8f acc = {};
#pragma unroll
        for (int k0 = 0; k0 < HID; k0 += 32) {
            v16bf a = make_frag(A + (size_t)arow_c * HID + k0 + koff, 1);
            // B[k][n] = Wt[n][k], Wt rows contiguous over K
            v16bf b = make_frag(Wt + (size_t)(ct * 16 + col) * HID + k0 + koff, 1);
            acc = __builtin_amdgcn_wmma_f32_16x16x32_bf16(
                false, a, false, b, (short)0, acc, false, false);
        }
        float* out = matsel ? gh : gi;
#pragma unroll
        for (int r = 0; r < 8; ++r)
            out[(r0 + r) * 192 + ct * 16 + col] = acc[r];
    }
    __syncthreads();

    // fused gate math: r,z,n and h update (torch GRUCell gate order r,z,n)
#pragma unroll
    for (int base = 0; base < 16 * HID; base += 256) {
        int idx  = base + threadIdx.x;
        int nloc = idx >> 6;
        int c    = idx & 63;
        int node = nodebase + nloc;
        if (node < N) {
            float ir = gi[nloc * 192 + c]        + b_ih[c];
            float iz = gi[nloc * 192 + 64 + c]   + b_ih[64 + c];
            float in_= gi[nloc * 192 + 128 + c]  + b_ih[128 + c];
            float hr = gh[nloc * 192 + c]        + b_hh[c];
            float hz = gh[nloc * 192 + 64 + c]   + b_hh[64 + c];
            float hn = gh[nloc * 192 + 128 + c]  + b_hh[128 + c];
            float r = 1.f / (1.f + __expf(-(ir + hr)));
            float z = 1.f / (1.f + __expf(-(iz + hz)));
            float nn = tanhf(in_ + r * hn);
            size_t off = (size_t)node * HID + c;
            h[off] = (1.f - z) * nn + z * h[off];
        }
    }
}

// ---------------------------------------------------------------------------
// Kernel 5: logits = h @ w2.T + b2 ; softmax over 40 classes.
// One thread per node; w2 staged in LDS via async global->LDS copy
// (gfx1250 GLOBAL_LOAD_ASYNC_TO_LDS_B128, tracked by ASYNCcnt).
// ---------------------------------------------------------------------------
__global__ __launch_bounds__(256)
void k_classify(const float* __restrict__ h, const float* __restrict__ w2,
                const float* __restrict__ b2, float* __restrict__ out, int N) {
    __shared__ float sw[NCLASS * HID];
    __shared__ float sb[NCLASS];

    // Async-stage w2 (10240 B = 640 x b128) into LDS.
    {
        unsigned long long gbase = (unsigned long long)(uintptr_t)w2;
        for (int i = threadIdx.x; i < (NCLASS * HID) / 4; i += 256) {
            unsigned lds_off = (unsigned)(uintptr_t)(&sw[i * 4]); // low 32 bits = LDS byte addr
            unsigned goff    = (unsigned)(i * 16);
            asm volatile("global_load_async_to_lds_b128 %0, %1, %2 offset:0"
                         :: "v"(lds_off), "v"(goff), "s"(gbase)
                         : "memory");
        }
    }
    if (threadIdx.x < NCLASS) sb[threadIdx.x] = b2[threadIdx.x];
    asm volatile("s_wait_asynccnt 0x0" ::: "memory");
    __syncthreads();

    int node = blockIdx.x * 256 + threadIdx.x;
    if (node >= N) return;

    float hv[HID];
    const float4* hp = (const float4*)(h + (size_t)node * HID);
#pragma unroll
    for (int i = 0; i < HID / 4; ++i) {
        float4 v = hp[i];
        hv[4 * i + 0] = v.x; hv[4 * i + 1] = v.y;
        hv[4 * i + 2] = v.z; hv[4 * i + 3] = v.w;
    }

    float logits[NCLASS];
    float mx = -1e30f;
#pragma unroll
    for (int c = 0; c < NCLASS; ++c) {
        float s = sb[c];
#pragma unroll 8
        for (int k = 0; k < HID; ++k) s += hv[k] * sw[c * HID + k];
        logits[c] = s;
        mx = fmaxf(mx, s);
    }
    float denom = 0.f;
#pragma unroll
    for (int c = 0; c < NCLASS; ++c) {
        float e = __expf(logits[c] - mx);
        logits[c] = e;
        denom += e;
    }
    float inv = 1.f / denom;
#pragma unroll
    for (int c = 0; c < NCLASS; ++c)
        out[(size_t)node * NCLASS + c] = logits[c] * inv;
}

// ---------------------------------------------------------------------------
extern "C" void kernel_launch(void* const* d_in, const int* in_sizes, int n_in,
                              void* d_out, int out_size, void* d_ws, size_t ws_size,
                              hipStream_t stream) {
    const float* x    = (const float*)d_in[0];
    const int*   ei   = (const int*)d_in[1];
    const float* w1   = (const float*)d_in[2];
    const float* b1   = (const float*)d_in[3];
    const float* gg_w = (const float*)d_in[4];
    const float* w_ih = (const float*)d_in[5];
    const float* w_hh = (const float*)d_in[6];
    const float* b_ih = (const float*)d_in[7];
    const float* b_hh = (const float*)d_in[8];
    const float* w2   = (const float*)d_in[9];
    const float* b2   = (const float*)d_in[10];

    const int N = in_sizes[0] / NFEAT;
    const int E = in_sizes[1] / 2;
    const int* src = ei;
    const int* dst = ei + E;
    float* out = (float*)d_out;

    const size_t nh = (size_t)N * HID;
    float* h   = (float*)d_ws;
    float* m   = h + nh;
    float* agg = m + nh;

    dim3 blk(256);
    const int gemm_blocks = (N + 127) / 128;

    k_input_gemm<<<gemm_blocks, blk, 0, stream>>>(x, w1, b1, h, N);

    for (int layer = 0; layer < 3; ++layer) {
        k_msg_gemm<<<gemm_blocks, blk, 0, stream>>>(h, gg_w + (size_t)layer * HID * HID, m, N);
        hipMemsetAsync(agg, 0, nh * sizeof(float), stream);
        long long total = (long long)E * HID;
        k_scatter<<<(unsigned)((total + 255) / 256), blk, 0, stream>>>(m, src, dst, agg, E);
        k_gru<<<(N + 15) / 16, blk, 0, stream>>>(agg, h, w_ih, w_hh, b_ih, b_hh, N);
    }

    k_classify<<<(N + 255) / 256, blk, 0, stream>>>(h, w2, b2, out, N);
}
